// SelfAttn_79748952752148
// MI455X (gfx1250) — compile-verified, matrix operand order checked by reading, not run on hardware
//
#include <hip/hip_runtime.h>

// ---------------------------------------------------------------------------
// SelfAttn (linear attention variant) for MI455X / gfx1250, wave32 + WMMA.
//   K1: QKV = Xt @ Wqkv^T        (bf16 WMMA 16x16x32, fp32 accum)
//   K2: per-token RoPE/softmax/ctx/o (fp32 WMMA 16x16x4, async LDS staging)
//   K3: out = Wlin @ Osp + bias  (bf16 WMMA 16x16x32, fp32 accum)
// Problem: b=64, c=1024, n=256 tokens, heads=16, d_head=64.
//
// LDS for the bf16 GEMMs is fragment-major: tile[row][kpair], row stride 68
// dwords (272B, 16B-aligned rows, 2-way bank conflicts only), so each WMMA
// operand fragment is two ds_load_b128 straight into the source registers.
// ---------------------------------------------------------------------------

typedef __attribute__((ext_vector_type(16))) __bf16    v16bf;
typedef __attribute__((ext_vector_type(8)))  float     v8f;
typedef __attribute__((ext_vector_type(8)))  unsigned  v8u;
typedef __attribute__((ext_vector_type(2)))  float     v2f;

#define B_DIM   64
#define C_DIM   1024
#define N_TOK   256        // tokens per batch image (16*16)
#define H_DIM   16         // heads
#define D_HEAD  64
#define QKV_COLS 3072
#define TOKENS  (B_DIM * N_TOK)      // 16384
#define OSP_PER_B (C_DIM * N_TOK)    // 262144 floats per batch

#define KC      128                  // K-chunk (64 kpairs)
#define LDS_STRIDE 68                // kpairs + pad: 272B rows, 16B aligned

#if __has_builtin(__builtin_amdgcn_global_load_async_to_lds_b128) && \
    __has_builtin(__builtin_amdgcn_s_wait_asynccnt)
#define USE_ASYNC_LDS 1
// Exact pointee type per hipcc diagnostic: GCC-vector int4, AS1 src / AS3 dst.
typedef int v4i_gcc __attribute__((vector_size(16)));
typedef __attribute__((address_space(1))) v4i_gcc* g4p;
typedef __attribute__((address_space(3))) v4i_gcc* l4p;
#else
#define USE_ASYNC_LDS 0
#endif

__device__ __forceinline__ unsigned f2bf(float f) {
    unsigned u = __builtin_bit_cast(unsigned, f);
    unsigned r = u + 0x7FFFu + ((u >> 16) & 1u);   // round-to-nearest-even
    return r >> 16;
}
__device__ __forceinline__ unsigned pack_bf2(float lo, float hi) {
    return f2bf(lo) | (f2bf(hi) << 16);
}
__device__ __forceinline__ v8f v8f_zero() {
    v8f z;
#pragma unroll
    for (int i = 0; i < 8; ++i) z[i] = 0.0f;
    return z;
}
// Load one 16-bit-element WMMA operand fragment (8 dwords = 16 bf16) from a
// fragment-major LDS tile row: kpairs [ks*16 + half*4 .. +3] and [+8 ..].
__device__ __forceinline__ v16bf load_frag(const unsigned* tileRow, int ks, int half) {
    const uint4 lo = *(const uint4*)(tileRow + ks * 16 + half * 4);
    const uint4 hi = *(const uint4*)(tileRow + ks * 16 + 8 + half * 4);
    v8u a;
    a[0] = lo.x; a[1] = lo.y; a[2] = lo.z; a[3] = lo.w;
    a[4] = hi.x; a[5] = hi.y; a[6] = hi.z; a[7] = hi.w;
    return __builtin_bit_cast(v16bf, a);
}

// ---------------------------------------------------------------------------
// Kernel 1: QKV[row, col] = sum_k Xt[row,k] * Wqkv[col,k]
//   row = b*256 + p ; Xt[row,k] = x[(b*1024 + k)*256 + p]
//   WG tile 64 rows x 128 cols, Kc = 128, 8 waves each compute 16x64.
// ---------------------------------------------------------------------------
__global__ __launch_bounds__(256) void k_qkv_gemm(const float* __restrict__ x,
                                                  const float* __restrict__ w_qkv,
                                                  float* __restrict__ qkv) {
    const int row0 = blockIdx.x * 64;          // 64-row tile never crosses b
    const int col0 = blockIdx.y * 128;
    const int b    = row0 >> 8;
    const int p0   = row0 & 255;
    const int tid  = threadIdx.x;
    const int lane = tid & 31;
    const int wave = tid >> 5;
    const int wr   = wave & 3;                 // row subtile (16 rows)
    const int wc   = wave >> 2;                // col subtile (64 cols)
    const int half = lane >> 4, l15 = lane & 15;

    __shared__ __align__(16) unsigned As2[64][LDS_STRIDE];    // [r][kpair]
    __shared__ __align__(16) unsigned Bs2[128][LDS_STRIDE];   // [n][kpair]

    v8f acc[4];
#pragma unroll
    for (int i = 0; i < 4; ++i) acc[i] = v8f_zero();

    for (int k0 = 0; k0 < C_DIM; k0 += KC) {
        // ---- stage A: x[(b*1024+k)*256 + p0 + r] (contiguous r, float4)
#pragma unroll
        for (int it = 0; it < 4; ++it) {
            int idx = tid + it * 256;          // 1024 = 64 kpairs * 16 r-quads
            int kp = idx >> 4, rq = (idx & 15) * 4;
            const float* src = x + ((size_t)(b * C_DIM + k0 + 2 * kp) * N_TOK + p0 + rq);
            float4 f0 = *(const float4*)src;
            float4 f1 = *(const float4*)(src + N_TOK);
            As2[rq + 0][kp] = pack_bf2(f0.x, f1.x);
            As2[rq + 1][kp] = pack_bf2(f0.y, f1.y);
            As2[rq + 2][kp] = pack_bf2(f0.z, f1.z);
            As2[rq + 3][kp] = pack_bf2(f0.w, f1.w);
            if (k0 + KC < C_DIM)
                __builtin_prefetch(src + KC * N_TOK, 0, 0);
        }
        // ---- stage B: w_qkv[(col0+n)*1024 + k] (contiguous k, float4 -> uint4)
        {
            int n  = tid & 127;
            int kq = tid >> 7;                  // 0..1 -> 64 consecutive k each
            const float4* wp = (const float4*)(w_qkv + (size_t)(col0 + n) * C_DIM + k0 + kq * 64);
#pragma unroll
            for (int t = 0; t < 4; ++t) {
                float4 f0 = wp[t * 4 + 0], f1 = wp[t * 4 + 1];
                float4 f2 = wp[t * 4 + 2], f3 = wp[t * 4 + 3];
                uint4 pk;
                pk.x = pack_bf2(f0.x, f0.y); pk.y = pack_bf2(f0.z, f0.w);
                pk.z = pack_bf2(f1.x, f1.y); pk.w = pack_bf2(f1.z, f1.w);
                *(uint4*)&Bs2[n][kq * 32 + t * 8] = pk;
                pk.x = pack_bf2(f2.x, f2.y); pk.y = pack_bf2(f2.z, f2.w);
                pk.z = pack_bf2(f3.x, f3.y); pk.w = pack_bf2(f3.z, f3.w);
                *(uint4*)&Bs2[n][kq * 32 + t * 8 + 4] = pk;
            }
            if (k0 + KC < C_DIM)
                __builtin_prefetch((const float*)wp + KC, 0, 0);
        }
        __syncthreads();

        // ---- 4 WMMA k-steps (K=32 each)
#pragma unroll
        for (int ks = 0; ks < 4; ++ks) {
            v16bf a = load_frag(As2[wr * 16 + l15], ks, half);
#pragma unroll
            for (int nt = 0; nt < 4; ++nt) {
                v16bf bb = load_frag(Bs2[wc * 64 + nt * 16 + l15], ks, half);
                acc[nt] = __builtin_amdgcn_wmma_f32_16x16x32_bf16(
                    false, a, false, bb, (short)0, acc[nt], false, false);
            }
        }
        __syncthreads();
    }

    // ---- store D tiles (fp32 QKV intermediate)
#pragma unroll
    for (int nt = 0; nt < 4; ++nt) {
#pragma unroll
        for (int r = 0; r < 8; ++r) {
            int row = row0 + wr * 16 + r + 8 * half;
            int col = col0 + wc * 64 + nt * 16 + l15;
            qkv[(size_t)row * QKV_COLS + col] = acc[nt][r];
        }
    }
}

// ---------------------------------------------------------------------------
// Kernel 2: one wave (32 threads) per token.
//   RoPE(q,k); q=softmax_d(q)*d^-0.5; k=softmax_h(k);
//   ctx = k^T v (64x64); o = q ctx (16x64);
//   store Osp[b][n*1024 + e*16 + h]  (== reference's scrambled reshape)
// ---------------------------------------------------------------------------
__global__ __launch_bounds__(32) void k_attn(const float* __restrict__ qkv,
                                             float* __restrict__ osp) {
    const int tok = blockIdx.x;
    const int b   = tok >> 8;
    const int n   = tok & 255;
    const int L    = threadIdx.x;
    const int half = L >> 4, l15 = L & 15;

    __shared__ __align__(16) float q_s[H_DIM * D_HEAD];   // [h][d]
    __shared__ __align__(16) float k_s[H_DIM * D_HEAD];
    __shared__ __align__(16) float v_s[H_DIM * D_HEAD];
    __shared__ __align__(16) float ctx[D_HEAD * D_HEAD];  // [d][e]

    const float* row = qkv + (size_t)tok * QKV_COLS;

#if USE_ASYNC_LDS
    // Pure fp32 copy global->LDS: use CDNA5 async-to-LDS path (ASYNCcnt).
#pragma unroll
    for (int it = 0; it < 8; ++it) {
        int i4 = L + it * 32;                  // float4 index, 256 per matrix
        int h = i4 >> 4, d = (i4 & 15) * 4;
        float* s = const_cast<float*>(row + h * 192 + d);
        __builtin_amdgcn_global_load_async_to_lds_b128((g4p)(s),       (l4p)(q_s + h * 64 + d), 0, 0);
        __builtin_amdgcn_global_load_async_to_lds_b128((g4p)(s + 64),  (l4p)(k_s + h * 64 + d), 0, 0);
        __builtin_amdgcn_global_load_async_to_lds_b128((g4p)(s + 128), (l4p)(v_s + h * 64 + d), 0, 0);
    }
    __builtin_amdgcn_s_wait_asynccnt(0);
#else
#pragma unroll
    for (int it = 0; it < 32; ++it) {          // 1024 elems / 32 lanes
        int i = L + it * 32;
        int h = i >> 6, d = i & 63;
        const float* hp = row + h * 192 + d;
        q_s[i] = hp[0];
        k_s[i] = hp[64];
        v_s[i] = hp[128];
    }
#endif
    __syncthreads();

    // RoPE (rotate-half), angle = n / 10000^(d/32), d < 32
    const float NEG_L2_1E4_OVER_32 = -13.287712379549449f / 32.0f;
#pragma unroll
    for (int it = 0; it < 16; ++it) {          // 512 (h,d) pairs
        int i = L + it * 32;
        int h = i >> 5, d = i & 31;
        float ang = (float)n * exp2f((float)d * NEG_L2_1E4_OVER_32);
        float cs = __cosf(ang), sn = __sinf(ang);
        int i1 = h * 64 + d, i2 = i1 + 32;
        float q1 = q_s[i1], q2 = q_s[i2];
        q_s[i1] = q1 * cs - q2 * sn;  q_s[i2] = q2 * cs + q1 * sn;
        float k1 = k_s[i1], k2 = k_s[i2];
        k_s[i1] = k1 * cs - k2 * sn;  k_s[i2] = k2 * cs + k1 * sn;
    }
    __syncthreads();

    // q: softmax over d per head, * 64^-0.5   (lanes 0..15, one head each)
    if (L < H_DIM) {
        float m = -3.4e38f;
        for (int d = 0; d < 64; ++d) m = fmaxf(m, q_s[L * 64 + d]);
        float s = 0.f;
        for (int d = 0; d < 64; ++d) { float e = __expf(q_s[L * 64 + d] - m); q_s[L * 64 + d] = e; s += e; }
        float inv = 0.125f / s;
        for (int d = 0; d < 64; ++d) q_s[L * 64 + d] *= inv;
    }
    // k: softmax over heads per d  (64 columns over 32 lanes)
    for (int d = L; d < 64; d += 32) {
        float m = -3.4e38f;
        for (int h = 0; h < 16; ++h) m = fmaxf(m, k_s[h * 64 + d]);
        float s = 0.f;
        for (int h = 0; h < 16; ++h) { float e = __expf(k_s[h * 64 + d] - m); k_s[h * 64 + d] = e; s += e; }
        float inv = 1.0f / s;
        for (int h = 0; h < 16; ++h) k_s[h * 64 + d] *= inv;
    }
    __syncthreads();

    // ctx[d][e] = sum_h k[h][d] * v[h][e] : 4x4 tiles, K=16 in 4 steps of 4
    v8f acc[4][4];
#pragma unroll
    for (int i = 0; i < 4; ++i)
#pragma unroll
        for (int j = 0; j < 4; ++j) acc[i][j] = v8f_zero();

#pragma unroll
    for (int ks = 0; ks < 4; ++ks) {
        v2f a[4], bb[4];
#pragma unroll
        for (int mt = 0; mt < 4; ++mt)
#pragma unroll
            for (int j = 0; j < 2; ++j)
                a[mt][j] = k_s[(ks * 4 + 2 * half + j) * 64 + mt * 16 + l15];  // A = k^T
#pragma unroll
        for (int nt = 0; nt < 4; ++nt)
#pragma unroll
            for (int j = 0; j < 2; ++j)
                bb[nt][j] = v_s[(ks * 4 + 2 * half + j) * 64 + nt * 16 + l15]; // B = v
#pragma unroll
        for (int mt = 0; mt < 4; ++mt)
#pragma unroll
            for (int nt = 0; nt < 4; ++nt)
                acc[mt][nt] = __builtin_amdgcn_wmma_f32_16x16x4_f32(
                    false, a[mt], false, bb[nt], (short)0, acc[mt][nt], false, false);
    }
    __syncthreads();
#pragma unroll
    for (int mt = 0; mt < 4; ++mt)
#pragma unroll
        for (int nt = 0; nt < 4; ++nt)
#pragma unroll
            for (int r = 0; r < 8; ++r)
                ctx[(mt * 16 + r + 8 * half) * 64 + nt * 16 + l15] = acc[mt][nt][r];
    __syncthreads();

    // o[h][e] = sum_d q[h][d] * ctx[d][e] : 1x4 tiles, K=64 in 16 steps
    v8f oacc[4];
#pragma unroll
    for (int i = 0; i < 4; ++i) oacc[i] = v8f_zero();
#pragma unroll
    for (int ks = 0; ks < 16; ++ks) {
        v2f a;
#pragma unroll
        for (int j = 0; j < 2; ++j)
            a[j] = q_s[l15 * 64 + ks * 4 + 2 * half + j];
#pragma unroll
        for (int nt = 0; nt < 4; ++nt) {
            v2f bb;
#pragma unroll
            for (int j = 0; j < 2; ++j)
                bb[j] = ctx[(ks * 4 + 2 * half + j) * 64 + nt * 16 + l15];
            oacc[nt] = __builtin_amdgcn_wmma_f32_16x16x4_f32(
                false, a, false, bb, (short)0, oacc[nt], false, false);
        }
    }

    // scrambled reshape store: Osp[b][n*1024 + e*16 + h]
    float* obase = osp + (size_t)b * OSP_PER_B + (size_t)n * 1024;
#pragma unroll
    for (int nt = 0; nt < 4; ++nt)
#pragma unroll
        for (int r = 0; r < 8; ++r) {
            int e = nt * 16 + l15;
            int h = r + 8 * half;
            obase[e * 16 + h] = oacc[nt][r];
        }
}

// ---------------------------------------------------------------------------
// Kernel 3: out[b,o,p] = sum_c Wlin[o,c] * Osp[b][c*256+p] + bias[o]
//   per-batch GEMM M=1024, N=256, K=1024; WG tile 64x128, Kc=128.
// ---------------------------------------------------------------------------
__global__ __launch_bounds__(256) void k_out_gemm(const float* __restrict__ osp,
                                                  const float* __restrict__ w_lin,
                                                  const float* __restrict__ b_lin,
                                                  float* __restrict__ out) {
    const int m0 = blockIdx.x * 64;
    const int n0 = blockIdx.y * 128;
    const int b  = blockIdx.z;
    const int tid  = threadIdx.x;
    const int lane = tid & 31;
    const int wave = tid >> 5;
    const int wr = wave & 3, wc = wave >> 2;
    const int half = lane >> 4, l15 = lane & 15;

    const float* ob = osp + (size_t)b * OSP_PER_B;

    __shared__ __align__(16) unsigned As2[64][LDS_STRIDE];    // [r][kpair] : Wlin rows
    __shared__ __align__(16) unsigned Bs2[128][LDS_STRIDE];   // [n][kpair] : Osp cols

    v8f acc[4];
#pragma unroll
    for (int i = 0; i < 4; ++i) acc[i] = v8f_zero();

    for (int k0 = 0; k0 < C_DIM; k0 += KC) {
        // ---- stage A: w_lin[(m0+r)*1024 + k]  (contiguous k, float4 -> uint4)
        {
            int r  = tid & 63;
            int kq = tid >> 6;                  // 0..3 -> 32 consecutive k each
            const float4* ap = (const float4*)(w_lin + (size_t)(m0 + r) * C_DIM + k0 + kq * 32);
#pragma unroll
            for (int t = 0; t < 2; ++t) {
                float4 f0 = ap[t * 4 + 0], f1 = ap[t * 4 + 1];
                float4 f2 = ap[t * 4 + 2], f3 = ap[t * 4 + 3];
                uint4 pk;
                pk.x = pack_bf2(f0.x, f0.y); pk.y = pack_bf2(f0.z, f0.w);
                pk.z = pack_bf2(f1.x, f1.y); pk.w = pack_bf2(f1.z, f1.w);
                *(uint4*)&As2[r][kq * 16 + t * 8] = pk;
                pk.x = pack_bf2(f2.x, f2.y); pk.y = pack_bf2(f2.z, f2.w);
                pk.z = pack_bf2(f3.x, f3.y); pk.w = pack_bf2(f3.z, f3.w);
                *(uint4*)&As2[r][kq * 16 + t * 8 + 4] = pk;
            }
            if (k0 + KC < C_DIM)
                __builtin_prefetch((const float*)ap + KC, 0, 0);
        }
        // ---- stage B: Osp[(k0+kk)*256 + n0 + n]  (contiguous n, float4)
#pragma unroll
        for (int it = 0; it < 8; ++it) {
            int idx = tid + it * 256;           // 2048 = 64 kpairs * 32 n-quads
            int kp = idx >> 5, nq = (idx & 31) * 4;
            const float* src = ob + (size_t)(k0 + 2 * kp) * N_TOK + n0 + nq;
            float4 f0 = *(const float4*)src;
            float4 f1 = *(const float4*)(src + N_TOK);
            Bs2[nq + 0][kp] = pack_bf2(f0.x, f1.x);
            Bs2[nq + 1][kp] = pack_bf2(f0.y, f1.y);
            Bs2[nq + 2][kp] = pack_bf2(f0.z, f1.z);
            Bs2[nq + 3][kp] = pack_bf2(f0.w, f1.w);
            if (k0 + KC < C_DIM)
                __builtin_prefetch(src + KC * N_TOK, 0, 0);
        }
        __syncthreads();

#pragma unroll
        for (int ks = 0; ks < 4; ++ks) {
            v16bf a = load_frag(As2[wr * 16 + l15], ks, half);
#pragma unroll
            for (int nt = 0; nt < 4; ++nt) {
                v16bf bb = load_frag(Bs2[wc * 64 + nt * 16 + l15], ks, half);
                acc[nt] = __builtin_amdgcn_wmma_f32_16x16x32_bf16(
                    false, a, false, bb, (short)0, acc[nt], false, false);
            }
        }
        __syncthreads();
    }

    // ---- epilogue: bias + store out[(b*1024 + m)*256 + n]
#pragma unroll
    for (int r = 0; r < 8; ++r) {
        int m = m0 + wr * 16 + r + 8 * half;
        float bias = b_lin[m];
#pragma unroll
        for (int nt = 0; nt < 4; ++nt) {
            int nn = n0 + wc * 64 + nt * 16 + l15;
            out[((size_t)b * C_DIM + m) * N_TOK + nn] = acc[nt][r] + bias;
        }
    }
}

// ---------------------------------------------------------------------------
extern "C" void kernel_launch(void* const* d_in, const int* in_sizes, int n_in,
                              void* d_out, int out_size, void* d_ws, size_t ws_size,
                              hipStream_t stream) {
    const float* x     = (const float*)d_in[0];   // [64,1024,16,16]
    const float* w_qkv = (const float*)d_in[1];   // [3072,1024]
    const float* w_lin = (const float*)d_in[2];   // [1024,1024,1,1]
    const float* b_lin = (const float*)d_in[3];   // [1024]
    // d_in[4] = n_heads (16), compile-time constant here.

    float* qkv = (float*)d_ws;                                       // 16384*3072 f32
    float* osp = (float*)((char*)d_ws + (size_t)TOKENS * QKV_COLS * sizeof(float));
    float* out = (float*)d_out;                                      // [64,1024,16,16]

    k_qkv_gemm<<<dim3(TOKENS / 64, QKV_COLS / 128), 256, 0, stream>>>(x, w_qkv, qkv);
    k_attn<<<dim3(TOKENS), 32, 0, stream>>>(qkv, osp);
    k_out_gemm<<<dim3(C_DIM / 64, N_TOK / 128, B_DIM), 256, 0, stream>>>(osp, w_lin, b_lin, out);
}
